// SuperGlue_25984552141180
// MI455X (gfx1250) — compile-verified, matrix operand order checked by reading, not run on hardware
//
#include <hip/hip_runtime.h>

typedef long long ll;
typedef __attribute__((ext_vector_type(16))) __bf16 v16bf;
typedef __attribute__((ext_vector_type(8)))  __bf16 v8bf;
typedef __attribute__((ext_vector_type(2)))  __bf16 v2bf;
typedef __attribute__((ext_vector_type(2)))  float  v2f;
typedef __attribute__((ext_vector_type(8)))  float  v8f;

#define TM 128
#define TN 128
#define TK 32

// ---------------------------------------------------------------- helpers
// pack two f32 -> dword of 2x bf16; vector fptrunc lets the backend pick the
// native packed conversion (v_cvt_*bf16*) on gfx1250.
__device__ __forceinline__ unsigned pack_bf16(float lo, float hi) {
  v2f f = {lo, hi};
  v2bf b = __builtin_convertvector(f, v2bf);
  return __builtin_bit_cast(unsigned, b);
}

__device__ __forceinline__ float blockMax(float v, float* red) {
  int tid = threadIdx.x;
  red[tid] = v; __syncthreads();
  for (int s = 128; s > 0; s >>= 1) {
    if (tid < s) red[tid] = fmaxf(red[tid], red[tid + s]);
    __syncthreads();
  }
  float r = red[0]; __syncthreads();
  return r;
}

__device__ __forceinline__ float blockSum(float v, float* red) {
  int tid = threadIdx.x;
  red[tid] = v; __syncthreads();
  for (int s = 128; s > 0; s >>= 1) {
    if (tid < s) red[tid] += red[tid + s];
    __syncthreads();
  }
  float r = red[0]; __syncthreads();
  return r;
}

// ---------------------------------------------------------------- WMMA GEMM
// C[z; m, n] = alpha * sum_k A[z; m, k] * X[z; k, n] (+ bias[m]) (+ Rres) (relu?)
// A element (m,k) at A[aoff + m*aRow + k*aCol], per-z offset (z/aDiv)*aSO + (z%aDiv)*aSI.
// Same scheme for X (k,n) and C (m,n).  bf16 inputs, f32 accumulate via
// v_wmma_f32_16x16x32_bf16.  Block = 256 thr = 8 waves (4x2), wave tile 32x64.
// Software pipeline: double-buffered LDS, next tile staged in VGPRs while the
// WMMAs consume the current tile; one barrier per K-step; packed bf16 LDS
// stores (b32); speculative prefetch 2 tiles ahead.
__global__ __launch_bounds__(256) void wmma_gemm(
    const float* __restrict__ A, ll aRow, ll aCol, int aDiv, ll aSO, ll aSI,
    const float* __restrict__ X, ll xRow, ll xCol, int xDiv, ll xSO, ll xSI,
    float* __restrict__ C, ll cRow, ll cCol, int cDiv, ll cSO, ll cSI,
    const float* __restrict__ bias, const float* __restrict__ Rres,
    float alpha, int relu, int M, int Ncols, int K)
{
  __shared__ __align__(16) __bf16 lA[2][TM][TK + 8];
  __shared__ __align__(16) __bf16 lB[2][TN][TK + 8];
  const int tid  = threadIdx.x;
  const int lane = tid & 31;
  const int wave = tid >> 5;
  const int wm = wave >> 1;     // 0..3
  const int wn = wave & 1;      // 0..1
  const int m0 = blockIdx.x * TM;
  const int n0 = blockIdx.y * TN;
  const int z  = blockIdx.z;
  const ll aoff = (ll)(z / aDiv) * aSO + (ll)(z % aDiv) * aSI;
  const ll xoff = (ll)(z / xDiv) * xSO + (ll)(z % xDiv) * xSI;
  const ll coff = (ll)(z / cDiv) * cSO + (ll)(z % cDiv) * cSI;
  const int half = lane >> 4;   // ISA: lanes 16-31 hold the second K half / M+8
  const int lx   = lane & 15;

  const v8f vzero = {0.f, 0.f, 0.f, 0.f, 0.f, 0.f, 0.f, 0.f};
  v8f acc[2][4];
#pragma unroll
  for (int i = 0; i < 2; ++i)
#pragma unroll
    for (int j = 0; j < 4; ++j) acc[i][j] = vzero;

  // tile staging: 8 k-adjacent pairs per thread per operand (128x32 / 256 thr)
  auto loadA = [&](int kk, float* ra) {
    if (aCol == 1) {            // row-major A: pair is contiguous 8B, coalesce k
#pragma unroll
      for (int j = 0; j < 8; ++j) {
        int pi = tid + 256 * j;
        int r = pi >> 4, cp = pi & 15;
        int gm = m0 + r, gk = kk + 2 * cp;
        bool ok = (gm < M) && (gk < K);
        const float* p = A + aoff + (ll)gm * aRow + gk;
        ra[2 * j]     = ok ? p[0] : 0.f;
        ra[2 * j + 1] = ok ? p[1] : 0.f;      // K is a multiple of TK
      }
    } else {                    // col-major A: coalesce along m
#pragma unroll
      for (int j = 0; j < 8; ++j) {
        int pi = tid + 256 * j;
        int r = pi & (TM - 1), cp = pi >> 7;
        int gm = m0 + r, gk = kk + 2 * cp;
        bool ok = (gm < M) && (gk < K);
        const float* p = A + aoff + (ll)gm * aRow + (ll)gk * aCol;
        ra[2 * j]     = ok ? p[0] : 0.f;
        ra[2 * j + 1] = ok ? p[aCol] : 0.f;
      }
    }
  };
  auto storeA = [&](int buf, const float* ra) {
    if (aCol == 1) {
#pragma unroll
      for (int j = 0; j < 8; ++j) {
        int pi = tid + 256 * j;
        *(unsigned*)&lA[buf][pi >> 4][2 * (pi & 15)] = pack_bf16(ra[2 * j], ra[2 * j + 1]);
      }
    } else {
#pragma unroll
      for (int j = 0; j < 8; ++j) {
        int pi = tid + 256 * j;
        *(unsigned*)&lA[buf][pi & (TM - 1)][2 * (pi >> 7)] = pack_bf16(ra[2 * j], ra[2 * j + 1]);
      }
    }
  };
  auto loadX = [&](int kk, float* rx) {
    if (xCol == 1) {            // coalesce along n; pair = two row loads
#pragma unroll
      for (int j = 0; j < 8; ++j) {
        int pi = tid + 256 * j;
        int n = pi & (TN - 1), cp = pi >> 7;
        int gn = n0 + n, gk = kk + 2 * cp;
        bool ok = (gn < Ncols) && (gk < K);
        const float* p = X + xoff + (ll)gk * xRow + gn;
        rx[2 * j]     = ok ? p[0] : 0.f;
        rx[2 * j + 1] = ok ? p[xRow] : 0.f;
      }
    } else {                    // coalesce along k (e.g. xRow==1): contiguous pair
#pragma unroll
      for (int j = 0; j < 8; ++j) {
        int pi = tid + 256 * j;
        int n = pi >> 4, cp = pi & 15;
        int gn = n0 + n, gk = kk + 2 * cp;
        bool ok = (gn < Ncols) && (gk < K);
        const float* p = X + xoff + (ll)gk * xRow + (ll)gn * xCol;
        rx[2 * j]     = ok ? p[0] : 0.f;
        rx[2 * j + 1] = ok ? p[xRow] : 0.f;
      }
    }
  };
  auto storeX = [&](int buf, const float* rx) {
    if (xCol == 1) {
#pragma unroll
      for (int j = 0; j < 8; ++j) {
        int pi = tid + 256 * j;
        *(unsigned*)&lB[buf][pi & (TN - 1)][2 * (pi >> 7)] = pack_bf16(rx[2 * j], rx[2 * j + 1]);
      }
    } else {
#pragma unroll
      for (int j = 0; j < 8; ++j) {
        int pi = tid + 256 * j;
        *(unsigned*)&lB[buf][pi >> 4][2 * (pi & 15)] = pack_bf16(rx[2 * j], rx[2 * j + 1]);
      }
    }
  };

  float ra[16], rx[16];
  loadA(0, ra);
  loadX(0, rx);
  int buf = 0;

  for (int kk = 0; kk < K; kk += TK) {
    storeA(buf, ra);
    storeX(buf, rx);
    __syncthreads();   // single barrier per K-step (also fences reuse of buf^1)

    // stage next tile while WMMAs run on the current one
    const int kn = kk + TK;
    if (kn < K) {
      loadA(kn, ra);
      loadX(kn, rx);
    }
    // speculative prefetch two tiles ahead -> global_prefetch_b8
    const int kp = kk + 2 * TK;
    if (kp < K) {
      int r = tid >> 5, c = tid & 31;
      int gm = m0 + r, gk = kp + c;
      if (gm < M && gk < K) __builtin_prefetch(&A[aoff + (ll)gm * aRow + (ll)gk * aCol]);
      int n = tid & (TN - 1), c2 = tid >> 7;
      int gn = n0 + n, gk2 = kp + c2;
      if (gn < Ncols && gk2 < K) __builtin_prefetch(&X[xoff + (ll)gk2 * xRow + (ll)gn * xCol]);
    }

    // Fragment layout (ISA 16-bit A 16x32): lane = 16*half + m,
    // element e: k = (e/8)*16 + half*8 + (e%8)  -> two b128 LDS loads.
    v16bf af[2], bfr[4];
#pragma unroll
    for (int fm = 0; fm < 2; ++fm) {
      int r = wm * 32 + fm * 16 + lx;
      v8bf x0 = *(const v8bf*)&lA[buf][r][half * 8];
      v8bf x1 = *(const v8bf*)&lA[buf][r][16 + half * 8];
      af[fm] = __builtin_shufflevector(x0, x1, 0,1,2,3,4,5,6,7,8,9,10,11,12,13,14,15);
    }
#pragma unroll
    for (int fn = 0; fn < 4; ++fn) {
      int r = wn * 64 + fn * 16 + lx;
      v8bf x0 = *(const v8bf*)&lB[buf][r][half * 8];
      v8bf x1 = *(const v8bf*)&lB[buf][r][16 + half * 8];
      bfr[fn] = __builtin_shufflevector(x0, x1, 0,1,2,3,4,5,6,7,8,9,10,11,12,13,14,15);
    }
#pragma unroll
    for (int fm = 0; fm < 2; ++fm)
#pragma unroll
      for (int fn = 0; fn < 4; ++fn)
        acc[fm][fn] = __builtin_amdgcn_wmma_f32_16x16x32_bf16(
            false, af[fm], false, bfr[fn], (short)0, acc[fm][fn], false, false);

    buf ^= 1;
  }

  // epilogue: C/D layout: VGPR r -> M = r + 8*half, lane%16 -> N
#pragma unroll
  for (int fm = 0; fm < 2; ++fm)
#pragma unroll
    for (int fn = 0; fn < 4; ++fn) {
      int n = n0 + wn * 64 + fn * 16 + lx;
#pragma unroll
      for (int r = 0; r < 8; ++r) {
        int m = m0 + wm * 32 + fm * 16 + r + 8 * half;
        if (m < M && n < Ncols) {
          ll off = coff + (ll)m * cRow + (ll)n * cCol;
          float v = acc[fm][fn][r] * alpha;
          if (bias) v += bias[m];
          if (Rres) v += Rres[off];
          if (relu) v = v > 0.f ? v : 0.f;
          C[off] = v;
        }
      }
    }
}

// ---------------------------------------------------------------- keypoint encoder
// One block per image set; BN stats over all B*N=2048 points via LDS atomics.
__global__ __launch_bounds__(256) void kenc_kernel(
    const float* __restrict__ k0, const float* __restrict__ k1,
    const float* __restrict__ s0, const float* __restrict__ s1,
    const float* __restrict__ d0, const float* __restrict__ d1,
    float* __restrict__ a0, float* __restrict__ a1,
    const float* w0, const float* b0, const float* g0, const float* be0,
    const float* w1, const float* b1, const float* g1, const float* be1,
    const float* w2, const float* b2, const float* g2, const float* be2,
    const float* w3, const float* b3, float* __restrict__ scratch)
{
  const int set = blockIdx.x;
  const float* kp = set ? k1 : k0;
  const float* sc = set ? s1 : s0;
  const float* de = set ? d1 : d0;
  float* ac = set ? a1 : a0;
  const int P = 2048;
  float* bufA = scratch + (ll)set * 2 * 256 * P;
  float* bufB = bufA + 256 * P;
  __shared__ float ssum[256], ssq[256], sscale[256], sshift[256];
  const int tid = threadIdx.x;

  for (int p = tid; p < P; p += 256) {
    float kx = kp[p * 2 + 0], ky = kp[p * 2 + 1];
    bufA[0 * P + p] = (kx - 320.f) / 448.f;   // (x - W/2) / (0.7*max(W,H))
    bufA[1 * P + p] = (ky - 240.f) / 448.f;
    bufA[2 * P + p] = sc[p];
  }
  __syncthreads();

  const float* Ws[4]  = {w0, w1, w2, w3};
  const float* Bs[4]  = {b0, b1, b2, b3};
  const float* Gs[3]  = {g0, g1, g2};
  const float* BEs[3] = {be0, be1, be2};
  const int dims[5] = {3, 32, 64, 128, 256};
  float* in = bufA; float* out = bufB;

  for (int l = 0; l < 4; ++l) {
    const int ci = dims[l], co = dims[l + 1];
    for (int c = tid; c < co; c += 256) { ssum[c] = 0.f; ssq[c] = 0.f; }
    __syncthreads();
    for (int i = tid; i < co * P; i += 256) {
      int c = i / P, p = i - c * P;
      float acc = Bs[l][c];
      const float* wr = Ws[l] + (ll)c * ci;
      for (int j = 0; j < ci; ++j) acc += wr[j] * in[j * P + p];
      if (l < 3) {
        out[i] = acc;
        atomicAdd(&ssum[c], acc);
        atomicAdd(&ssq[c], acc * acc);
      } else {
        int b = p >> 10, n = p & 1023;
        ll off = (ll)b * 256 * 1024 + (ll)c * 1024 + n;
        ac[off] = de[off] + acc;               // desc + kenc(kpts, scores)
      }
    }
    __syncthreads();
    if (l < 3) {
      for (int c = tid; c < co; c += 256) {
        float mean = ssum[c] / (float)P;
        float var  = ssq[c] / (float)P - mean * mean;
        float scl  = Gs[l][c] * rsqrtf(var + 1e-5f);
        sscale[c] = scl;
        sshift[c] = BEs[l][c] - mean * scl;
      }
      __syncthreads();
      for (int i = tid; i < co * P; i += 256) {
        int c = i / P;
        float v = out[i] * sscale[c] + sshift[c];
        out[i] = v > 0.f ? v : 0.f;
      }
      __syncthreads();
      float* t = in; in = out; out = t;
    }
  }
}

// ---------------------------------------------------------------- elementwise kernels
__global__ __launch_bounds__(256) void softmax_rows(float* __restrict__ S, int len) {
  __shared__ float red[256];
  const ll base = (ll)blockIdx.x * len;
  const int tid = threadIdx.x;
  float m = -3.4e38f;
  for (int j = tid; j < len; j += 256) m = fmaxf(m, S[base + j]);
  m = blockMax(m, red);
  float s = 0.f;
  for (int j = tid; j < len; j += 256) {
    float e = __expf(S[base + j] - m);
    S[base + j] = e; s += e;
  }
  s = blockSum(s, red);
  float inv = 1.f / s;
  for (int j = tid; j < len; j += 256) S[base + j] *= inv;
}

__global__ __launch_bounds__(256) void bn_relu(float* __restrict__ Y,
                                               const float* __restrict__ g,
                                               const float* __restrict__ be,
                                               int Cch, int Nn, int Bn) {
  __shared__ float red[256];
  const int c = blockIdx.x, tid = threadIdx.x;
  const int tot = Bn * Nn;
  float s = 0.f, s2 = 0.f;
  for (int i = tid; i < tot; i += 256) {
    int b = i / Nn, n = i - b * Nn;
    float v = Y[(ll)b * Cch * Nn + (ll)c * Nn + n];
    s += v; s2 += v * v;
  }
  s  = blockSum(s, red);
  s2 = blockSum(s2, red);
  float mean = s / (float)tot;
  float var  = s2 / (float)tot - mean * mean;
  float scl  = g[c] * rsqrtf(var + 1e-5f);
  float sft  = be[c] - mean * scl;
  for (int i = tid; i < tot; i += 256) {
    int b = i / Nn, n = i - b * Nn;
    ll off = (ll)b * Cch * Nn + (ll)c * Nn + n;
    float v = Y[off] * scl + sft;
    Y[off] = v > 0.f ? v : 0.f;
  }
}

__global__ __launch_bounds__(256) void copy_rows(float* __restrict__ dst,
                                                 const float* __restrict__ src,
                                                 ll dstStride, ll srcStride, int count) {
  int b = blockIdx.y;
  int i = blockIdx.x * 256 + threadIdx.x;
  if (i < count) dst[(ll)b * dstStride + i] = src[(ll)b * srcStride + i];
}

// ---------------------------------------------------------------- Sinkhorn
#define NP1 1025
__global__ __launch_bounds__(256) void ot_init(float* __restrict__ Z,
                                               float* __restrict__ u, float* __restrict__ v,
                                               const float* __restrict__ alpha_p) {
  float a = *alpha_p;
  int i = blockIdx.x * 256 + threadIdx.x;
  if (i < 2 * NP1) {
    int b = i / NP1, r = i - b * NP1;
    float* Zb = Z + (ll)b * NP1 * NP1;
    Zb[(ll)r * NP1 + 1024] = a;        // bins0 column
    Zb[(ll)1024 * NP1 + r] = a;        // bins1 row (+ corner)
    u[i] = 0.f; v[i] = 0.f;
  }
}

__device__ __forceinline__ float log_marg(int i) {
  float norm = -__logf(2048.f);
  return (i < 1024) ? norm : (__logf(1024.f) + norm);
}

__global__ __launch_bounds__(256) void sink_row(const float* __restrict__ Z,
                                                float* __restrict__ u,
                                                const float* __restrict__ v) {
  __shared__ float red[256];
  const int i = blockIdx.x, b = blockIdx.y, tid = threadIdx.x;
  const float* zr = Z + ((ll)b * NP1 + i) * NP1;
  const float* vb = v + b * NP1;
  float m = -3.4e38f;
  for (int j = tid; j < NP1; j += 256) m = fmaxf(m, zr[j] + vb[j]);
  m = blockMax(m, red);
  float s = 0.f;
  for (int j = tid; j < NP1; j += 256) s += __expf(zr[j] + vb[j] - m);
  s = blockSum(s, red);
  if (tid == 0) u[b * NP1 + i] = log_marg(i) - (m + __logf(s));
}

__global__ __launch_bounds__(256) void sink_col(const float* __restrict__ Z,
                                                float* __restrict__ v,
                                                const float* __restrict__ u) {
  __shared__ float red[256];
  const int j = blockIdx.x, b = blockIdx.y, tid = threadIdx.x;
  const float* zb = Z + (ll)b * NP1 * NP1;
  const float* ub = u + b * NP1;
  float m = -3.4e38f;
  for (int i = tid; i < NP1; i += 256) m = fmaxf(m, zb[(ll)i * NP1 + j] + ub[i]);
  m = blockMax(m, red);
  float s = 0.f;
  for (int i = tid; i < NP1; i += 256) s += __expf(zb[(ll)i * NP1 + j] + ub[i] - m);
  s = blockSum(s, red);
  if (tid == 0) v[b * NP1 + j] = log_marg(j) - (m + __logf(s));
}

__global__ __launch_bounds__(256) void ot_final(const float* __restrict__ Z,
                                                const float* __restrict__ u,
                                                const float* __restrict__ v,
                                                float* __restrict__ out) {
  int idx = blockIdx.x * 256 + threadIdx.x;
  const int tot = 2 * NP1 * NP1;
  if (idx < tot) {
    int b = idx / (NP1 * NP1);
    int rem = idx - b * NP1 * NP1;
    int i = rem / NP1, j = rem - i * NP1;
    out[idx] = Z[idx] + u[b * NP1 + i] + v[b * NP1 + j] + __logf(2048.f);
  }
}

// ---------------------------------------------------------------- host side
static inline void gemm(hipStream_t s,
    const float* A, ll aRow, ll aCol, int aDiv, ll aSO, ll aSI,
    const float* X, ll xRow, ll xCol, int xDiv, ll xSO, ll xSI,
    float* C, ll cRow, ll cCol, int cDiv, ll cSO, ll cSI,
    const float* bias, const float* R,
    float alpha, int relu, int M, int N, int K, int Z)
{
  dim3 g((M + TM - 1) / TM, (N + TN - 1) / TN, Z);
  wmma_gemm<<<g, dim3(256), 0, s>>>(A, aRow, aCol, aDiv, aSO, aSI,
                                    X, xRow, xCol, xDiv, xSO, xSI,
                                    C, cRow, cCol, cDiv, cSO, cSI,
                                    bias, R, alpha, relu, M, N, K);
}

extern "C" void kernel_launch(void* const* d_in, const int* in_sizes, int n_in,
                              void* d_out, int out_size, void* d_ws, size_t ws_size,
                              hipStream_t stream) {
  (void)in_sizes; (void)n_in; (void)out_size; (void)ws_size;
  const int B = 2, N = 1024, D = 256, L = 18, H = 4;
  const ll DN = (ll)D * N;            // 262144 per batch
  const ll ACT = (ll)B * DN;          // 524288

  const float* kpts0 = (const float*)d_in[0];
  const float* kpts1 = (const float*)d_in[1];
  const float* sc0   = (const float*)d_in[2];
  const float* sc1   = (const float*)d_in[3];
  const float* desc0 = (const float*)d_in[4];
  const float* desc1 = (const float*)d_in[5];
  const float* kw0 = (const float*)d_in[6],  *kb0 = (const float*)d_in[7];
  const float* kg0 = (const float*)d_in[8],  *kbe0 = (const float*)d_in[9];
  const float* kw1 = (const float*)d_in[10], *kb1 = (const float*)d_in[11];
  const float* kg1 = (const float*)d_in[12], *kbe1 = (const float*)d_in[13];
  const float* kw2 = (const float*)d_in[14], *kb2 = (const float*)d_in[15];
  const float* kg2 = (const float*)d_in[16], *kbe2 = (const float*)d_in[17];
  const float* kw3 = (const float*)d_in[18], *kb3 = (const float*)d_in[19];
  const float* proj_w  = (const float*)d_in[20];
  const float* proj_b  = (const float*)d_in[21];
  const float* merge_w = (const float*)d_in[22];
  const float* merge_b = (const float*)d_in[23];
  const float* mlp1_w  = (const float*)d_in[24];
  const float* mlp1_b  = (const float*)d_in[25];
  const float* bn_gp   = (const float*)d_in[26];
  const float* bn_bp   = (const float*)d_in[27];
  const float* mlp2_w  = (const float*)d_in[28];
  const float* mlp2_b  = (const float*)d_in[29];
  const float* final_w = (const float*)d_in[30];
  const float* final_b = (const float*)d_in[31];
  const float* bin_sc  = (const float*)d_in[32];

  float* ws = (float*)d_ws;
  float* act0 = ws;
  float* act1 = act0 + ACT;
  float* nac0 = act1 + ACT;
  float* nac1 = nac0 + ACT;
  float* Qb   = nac1 + ACT;
  float* Kb   = Qb + ACT;
  float* Vb   = Kb + ACT;
  float* msgb = Vb + ACT;
  float* ybuf = msgb + ACT;                 // [B][512][N]
  float* y1   = ybuf + 2 * ACT;             // [B][512][N]
  float* Sb   = y1 + 2 * ACT;               // [B*H][N][N] (also kenc scratch)
  float* coup = Sb + (ll)B * H * N * N;     // [B][1025][1025]
  float* ub   = coup + (ll)B * NP1 * NP1;
  float* vb   = ub + B * NP1;

  // 1) keypoint encoder + residual add into act0/act1
  kenc_kernel<<<dim3(2), dim3(256), 0, stream>>>(
      kpts0, kpts1, sc0, sc1, desc0, desc1, act0, act1,
      kw0, kb0, kg0, kbe0, kw1, kb1, kg1, kbe1,
      kw2, kb2, kg2, kbe2, kw3, kb3, Sb);

  float *cur0 = act0, *cur1 = act1, *nxt0 = nac0, *nxt1 = nac1;

  // 2) attentional GNN layers
  for (int l = 0; l < L; ++l) {
    const int cross = l & 1;
    const float* pw  = proj_w  + (ll)l * 3 * D * D;
    const float* pb  = proj_b  + (ll)l * 3 * D;
    const float* mw  = merge_w + (ll)l * D * D;
    const float* mb  = merge_b + (ll)l * D;
    const float* m1w = mlp1_w  + (ll)l * 512 * 512;
    const float* m1b = mlp1_b  + (ll)l * 512;
    const float* bg  = bn_gp   + (ll)l * 512;
    const float* bb  = bn_bp   + (ll)l * 512;
    const float* m2w = mlp2_w  + (ll)l * D * 512;
    const float* m2b = mlp2_b  + (ll)l * D;

    for (int t = 0; t < 2; ++t) {
      const float* x   = t ? cur1 : cur0;
      const float* src = t ? (cross ? cur0 : cur1) : (cross ? cur1 : cur0);
      float* outac = t ? nxt1 : nxt0;

      // Q = Wq x, K = Wk src, V = Wv src    ([D x D] x [D x N] per batch)
      gemm(stream, pw,           D, 1, 1, 0, 0,  x,   N, 1, 1, DN, 0,
           Qb, N, 1, 1, DN, 0, pb,       nullptr, 1.f, 0, D, N, D, B);
      gemm(stream, pw + D * D,   D, 1, 1, 0, 0,  src, N, 1, 1, DN, 0,
           Kb, N, 1, 1, DN, 0, pb + D,   nullptr, 1.f, 0, D, N, D, B);
      gemm(stream, pw + 2*D*D,   D, 1, 1, 0, 0,  src, N, 1, 1, DN, 0,
           Vb, N, 1, 1, DN, 0, pb + 2*D, nullptr, 1.f, 0, D, N, D, B);

      // scores[z=b*H+h][n][m] = (1/8) sum_d Q[b][d*H+h][n] K[b][d*H+h][m]
      gemm(stream, Qb, 1, (ll)H * N, H, DN, N,
                   Kb, (ll)H * N, 1, H, DN, N,
                   Sb, N, 1, 1, (ll)N * N, 0,
                   nullptr, nullptr, 0.125f, 0, N, N, D / H, B * H);
      softmax_rows<<<dim3(B * H * N), dim3(256), 0, stream>>>(Sb, N);

      // msg[b][d*H+h][n] = sum_m P[z][n][m] V[b][d*H+h][m]
      gemm(stream, Sb, N, 1, 1, (ll)N * N, 0,
                   Vb, 1, (ll)H * N, H, DN, N,
                   msgb, 1, (ll)H * N, H, DN, N,
                   nullptr, nullptr, 1.f, 0, N, D / H, N, B * H);

      // y = concat([x, merge(msg)])
      copy_rows<<<dim3((int)((DN + 255) / 256), B), dim3(256), 0, stream>>>(
          ybuf, x, (ll)2 * DN, DN, (int)DN);
      gemm(stream, mw, D, 1, 1, 0, 0,  msgb, N, 1, 1, DN, 0,
           ybuf + DN, N, 1, 1, (ll)2 * DN, 0, mb, nullptr, 1.f, 0, D, N, D, B);

      // mlp1 -> bn+relu -> mlp2 (+residual)
      gemm(stream, m1w, 512, 1, 1, 0, 0,  ybuf, N, 1, 1, (ll)2 * DN, 0,
           y1, N, 1, 1, (ll)2 * DN, 0, m1b, nullptr, 1.f, 0, 512, N, 512, B);
      bn_relu<<<dim3(512), dim3(256), 0, stream>>>(y1, bg, bb, 512, N, B);
      gemm(stream, m2w, 512, 1, 1, 0, 0,  y1, N, 1, 1, (ll)2 * DN, 0,
           outac, N, 1, 1, DN, 0, m2b, x, 1.f, 0, D, N, 512, B);
    }
    float* t0 = cur0; cur0 = nxt0; nxt0 = t0;
    float* t1 = cur1; cur1 = nxt1; nxt1 = t1;
  }

  // 3) final projections (reuse Qb/Kb as m0/m1)
  gemm(stream, final_w, D, 1, 1, 0, 0, cur0, N, 1, 1, DN, 0,
       Qb, N, 1, 1, DN, 0, final_b, nullptr, 1.f, 0, D, N, D, B);
  gemm(stream, final_w, D, 1, 1, 0, 0, cur1, N, 1, 1, DN, 0,
       Kb, N, 1, 1, DN, 0, final_b, nullptr, 1.f, 0, D, N, D, B);

  // scores[b][n][m] = (1/16) sum_d m0[b][d][n] m1[b][d][m]  -> coup[:, :N, :N]
  gemm(stream, Qb, 1, N, 1, DN, 0,
               Kb, N, 1, 1, DN, 0,
               coup, NP1, 1, 1, (ll)NP1 * NP1, 0,
               nullptr, nullptr, 1.f / 16.f, 0, N, N, D, B);

  // 4) Sinkhorn in log space (L2-resident)
  ot_init<<<dim3((2 * NP1 + 255) / 256), dim3(256), 0, stream>>>(coup, ub, vb, bin_sc);
  for (int it = 0; it < 50; ++it) {
    sink_row<<<dim3(NP1, B), dim3(256), 0, stream>>>(coup, ub, vb);
    sink_col<<<dim3(NP1, B), dim3(256), 0, stream>>>(coup, vb, ub);
  }
  ot_final<<<dim3((2 * NP1 * NP1 + 255) / 256), dim3(256), 0, stream>>>(
      coup, ub, vb, (float*)d_out);
}
